// transformer_33200097198366
// MI455X (gfx1250) — matrix operand order repaired and checked by reading, not compiled, not run on hardware
//
#include <hip/hip_runtime.h>
#include <hip/hip_bf16.h>

// MI455X (gfx1250, wave32) ensemble-attention block.
// Conv, QKV and attention-score GEMMs on v_wmma_f32_16x16x32_bf16 with
// vectorized (32B) LDS fragment loads in contiguous k order (k = 16*(lane/16) + s) for A and B.
// Pipeline is HBM-bound (~1.3 GB -> ~60us at 23.3 TB/s); all global access
// is coalesced b128, all WMMA operands are single 32B LDS vector loads
// except the inherent im2col gather of the conv A-tile.

#define HDIM 361
#define WDIM 720
#define HW   (HDIM * WDIM)      // 259920 (divisible by 16)
#define CIN  11
#define CEMB 32
#define NHEAD 16
#define NENS 10
#define KCONV 275               // 11 * 25
#define KPAD  288               // padded to 9 x 32
#define PROW 1120               // patch row: 1100 data + 16 zero pad + align

typedef __attribute__((ext_vector_type(8)))  float          v8f;
typedef __attribute__((ext_vector_type(16))) __bf16         v16bf;
typedef __attribute__((ext_vector_type(8)))  int            v8i;

// Involution permuting a 32-index: swaps groups 8..15 <-> 16..23.
// Maps "natural" order to the 16-bit A-matrix fragment order
// (lanes 0-15 read halves {0..7,16..23}, lanes 16-31 read {8..15,24..31}).
__device__ __forceinline__ int perm8(int w) {
  int g = (w >> 3) & 3;
  return w + ((g == 1) ? 8 : (g == 2) ? -8 : 0);
}

#define SELU_S  1.0507009873554805f
#define SELU_SA 1.7580993408473766f   // scale * alpha

// ---------------------------------------------------------------------------
// Kernel 0: zero the accumulators in workspace (harness poisons d_ws).
// ---------------------------------------------------------------------------
__global__ void init_kernel(float* __restrict__ stats, float* __restrict__ scores) {
  int t = blockIdx.x * blockDim.x + threadIdx.x;
  if (t < 32) stats[t] = 0.f;
  if (t < NHEAD * NENS * NENS) scores[t] = 0.f;
}

// ---------------------------------------------------------------------------
// Kernel 1: earth-padded 5x5 conv (implicit GEMM, bf16 WMMA) + SELU,
//           plus per-member LayerNorm sum / sum^2 accumulation.
// Block 256 = 8 waves; each wave: 16 row-consecutive pixels x 32 out channels.
// B fragments: contiguous v16bf loads from [co][k] weight matrix.
// A fragments: byte-offset-table gather (no div/mod, no u16 extracts);
// offsets vector-loaded as v8i (pre-scaled to bytes).
// ---------------------------------------------------------------------------
__global__ void __launch_bounds__(256) conv_emb_kernel(
    const float* __restrict__ inp, const float* __restrict__ emb_w,
    const float* __restrict__ emb_b, float* __restrict__ xbuf,
    float* __restrict__ stats)
{
  __shared__ __bf16 wldsT[CEMB * KPAD];       // [co][k] plain, 18 KB
  __shared__ __bf16 patch[8][PROW];           // per-wave 11x5x20 patch, 17.5 KB
  __shared__ int    aoff4[KPAD];              // k -> patch BYTE offset (zeros pad)
  __shared__ float  stage[8][CEMB * 16];      // transpose staging, 16 KB
  __shared__ float  s_sum, s_sumsq;

  const int tid = threadIdx.x;
  const int wav = tid >> 5, lane = tid & 31;
  const int e = blockIdx.z;

  if (tid == 0) { s_sum = 0.f; s_sumsq = 0.f; }
  // Weights as [co][k] bf16, zero-padded k 275..287.
  for (int idx = tid; idx < CEMB * KPAD; idx += 256) {
    int co = idx / KPAD, k = idx % KPAD;
    float w = 0.f;
    if (k < KCONV) {
      int ci = k / 25, rem = k % 25;
      w = emb_w[(co * CIN + ci) * 25 + rem];
    }
    wldsT[idx] = (__bf16)w;
  }
  // aoff4[k] = byte offset of (ci,ky,kx) in patch row; padding -> zero region.
  for (int k = tid; k < KPAD; k += 256) {
    int off = 1100;
    if (k < KCONV) {
      int ci = k / 25, rem = k % 25, ky = rem / 5, kx = rem % 5;
      off = ci * 100 + ky * 20 + kx;
    }
    aoff4[k] = off * 2;                       // pre-scaled: bf16 = 2 bytes
  }
  __syncthreads();

  const int tile = blockIdx.x * 8 + wav;
  const bool active = tile < HDIM * 45;
  float lsum = 0.f, lsumsq = 0.f;

  if (active) {
    const int y = tile / 45, x0 = (tile % 45) * 16;
    // 11x5x20 earth-padded patch (periodic lon, zero lat) as bf16.
    for (int idx = lane; idx < CIN * 100; idx += 32) {
      int ci = idx / 100, rem = idx % 100, row = rem / 20, col = rem % 20;
      int yy = y - 2 + row;
      int xx = x0 - 2 + col;
      xx += (xx < 0) ? WDIM : 0;
      xx -= (xx >= WDIM) ? WDIM : 0;
      float v = (yy >= 0 && yy < HDIM)
                  ? inp[((size_t)(e * CIN + ci) * HDIM + yy) * WDIM + xx] : 0.f;
      patch[wav][ci * 100 + row * 20 + col] = (__bf16)v;
    }
    if (lane < 20) patch[wav][1100 + lane] = (__bf16)0.f;   // zero-pad region

    const int m = lane & 15;
    const int l16 = lane & 16;
    const int hi8 = l16 >> 1;
    const __bf16* bptr0 = &wldsT[m * KPAD + l16];
    const __bf16* bptr1 = &wldsT[(m + 16) * KPAD + l16];
    const char* pbase = (const char*)&patch[wav][0] + 2 * m;
    v8f acc0 = {}, acc1 = {};

    #pragma unroll
    for (int ks = 0; ks < KPAD; ks += 32) {
      v16bf b0 = *(const v16bf*)(bptr0 + ks);               // 32B LDS load
      v16bf b1 = *(const v16bf*)(bptr1 + ks);               // 32B LDS load
      v8i off0 = *(const v8i*)&aoff4[ks + l16];             // 32B LDS load
      v8i off1 = *(const v8i*)&aoff4[ks + l16 + 8];         // 32B LDS load
      v16bf a;
      #pragma unroll
      for (int h = 0; h < 8; ++h) {
        a[h]     = *(const __bf16*)(pbase + off0[h]);
        a[h + 8] = *(const __bf16*)(pbase + off1[h]);
      }
      acc0 = __builtin_amdgcn_wmma_f32_16x16x32_bf16(false, a, false, b0,
                                                     (short)0, acc0, false, false);
      acc1 = __builtin_amdgcn_wmma_f32_16x16x32_bf16(false, a, false, b1,
                                                     (short)0, acc1, false, false);
    }

    const float bias0 = emb_b[m], bias1 = emb_b[16 + m];
    #pragma unroll
    for (int r = 0; r < 8; ++r) {
      float z0 = acc0[r] + bias0;
      float z1 = acc1[r] + bias1;
      z0 = (z0 > 0.f) ? SELU_S * z0 : SELU_SA * (__expf(z0) - 1.f);
      z1 = (z1 > 0.f) ? SELU_S * z1 : SELU_SA * (__expf(z1) - 1.f);
      int px = r + hi8;
      stage[wav][m * 16 + px] = z0;
      stage[wav][(m + 16) * 16 + px] = z1;
      lsum   += z0 + z1;
      lsumsq += z0 * z0 + z1 * z1;
    }
    // Coalesced store: lane = channel, 16 contiguous floats (4x b128).
    {
      const int ch = lane;
      const float4* src = (const float4*)&stage[wav][ch * 16];
      float4* dst = (float4*)&xbuf[((size_t)(e * CEMB + ch) * HDIM + y) * WDIM + x0];
      #pragma unroll
      for (int q = 0; q < 4; ++q) dst[q] = src[q];
    }
    atomicAdd(&s_sum, lsum);
    atomicAdd(&s_sumsq, lsumsq);
  }
  __syncthreads();
  if (tid == 0) {
    atomicAdd(&stats[e * 2 + 0], s_sum);
    atomicAdd(&stats[e * 2 + 1], s_sumsq);
  }
}

// ---------------------------------------------------------------------------
// Kernel 2: finalize LayerNorm statistics (mu, 1/sigma) per ensemble member.
// ---------------------------------------------------------------------------
__global__ void ln_stats_kernel(const float* __restrict__ stats,
                                float* __restrict__ murstd) {
  int t = threadIdx.x;
  if (t < NENS) {
    const float N = (float)CEMB * (float)HW;
    float s = stats[t * 2], ss = stats[t * 2 + 1];
    float mu = s / N;
    float var = ss / N - mu * mu;
    murstd[t * 2 + 0] = mu;
    murstd[t * 2 + 1] = rsqrtf(var + 1e-5f);
  }
}

// ---------------------------------------------------------------------------
// Kernel 3: LN + QKV projection (WMMA, K=32) for all 10 members on a 128-px
// block, then per-head ensemble score GEMM (16x16x128 WMMA over pixels).
// All WMMA operands are single 32B v16bf LDS loads:
//   xsT  : LN'd x tile, plain [px][c]  -> A operand
//   wqkvT: [n][c] projection matrix        -> B operands (hoisted)
//   kbuf : plain bf16 K           -> A operand of score GEMM
//   qbuf : plain bf16 Q                    -> B operand of score GEMM
// Next member's x rows are prefetched (global_prefetch_b8) to overlap HBM
// latency with the current member's WMMA + stores.
// ---------------------------------------------------------------------------
__global__ void __launch_bounds__(256) qkv_scores_kernel(
    const float* __restrict__ xbuf, const float* __restrict__ lnw,
    const float* __restrict__ lnb, const float* __restrict__ wvm,
    const float* __restrict__ wkm, const float* __restrict__ wqm,
    const float* __restrict__ murstd, float* __restrict__ vbuf,
    float* __restrict__ scores)
{
  extern __shared__ v16bf smem_v[];               // 32B-aligned base
  char* smem = (char*)smem_v;
  float*  lnw_s  = (float*)smem;                          // [32][128]
  float*  lnb_s  = lnw_s + CEMB * 128;                    // [32][128]
  __bf16* kbuf   = (__bf16*)(lnb_s + CEMB * 128);         // [10][16][128] plain
  __bf16* qbuf   = kbuf + NENS * NHEAD * 128;             // [10][16][128] plain
  __bf16* xsT    = qbuf + NENS * NHEAD * 128;             // [8][16 px][32 c]
  float*  vstage = (float*)(xsT + 8 * 16 * CEMB);         // [8][256]
  __bf16* wqkvT  = (__bf16*)(vstage + 8 * 256);           // [48][32]

  const int tid = threadIdx.x, wav = tid >> 5, lane = tid & 31;
  const int p0 = blockIdx.x * 128;

  for (int idx = tid; idx < CEMB * 128; idx += 256) {
    int c = idx >> 7, px = idx & 127;
    int p = p0 + px;
    lnw_s[idx] = (p < HW) ? lnw[(size_t)c * HW + p] : 0.f;
    lnb_s[idx] = (p < HW) ? lnb[(size_t)c * HW + p] : 0.f;
  }
  for (int idx = tid; idx < 48 * CEMB; idx += 256) {
    int n = idx >> 5, c = idx & 31;
    const float* wsrc = (n < 16) ? wvm : (n < 32) ? wkm : wqm;
    wqkvT[idx] = (__bf16)wsrc[(n & 15) * CEMB + c];
  }
  __syncthreads();

  const int m = lane & 15;
  const int l16 = lane & 16;
  const int hi8 = l16 >> 1;
  const int p0w = p0 + wav * 16;
  __bf16* xsT_w    = xsT + wav * (16 * CEMB);
  float*  vstage_w = vstage + wav * 256;

  // Member-invariant B fragments: one 32B load each, held in registers.
  const v16bf bv = *(const v16bf*)&wqkvT[m * 32 + l16];
  const v16bf bk = *(const v16bf*)&wqkvT[(16 + m) * 32 + l16];
  const v16bf bq = *(const v16bf*)&wqkvT[(32 + m) * 32 + l16];

  // Warm the first member's tile into cache.
  if (p0w < HW)
    __builtin_prefetch(&xbuf[(size_t)lane * HW + p0w], 0, 0);

  for (int ie = 0; ie < NENS; ++ie) {
    const float mu = murstd[ie * 2], rstd = murstd[ie * 2 + 1];
    // LN on the coalesced staging path; store plain bf16 [px][c].
    {
      const int c = lane;
      const int pc = perm8(c);
      const float* src = &xbuf[(size_t)(ie * CEMB + c) * HW + p0w];
      const float* lwp = &lnw_s[c * 128 + wav * 16];
      const float* lbp = &lnb_s[c * 128 + wav * 16];
      #pragma unroll
      for (int q = 0; q < 16; ++q) {
        float xv = (p0w + q < HW) ? src[q] : 0.f;
        xsT_w[q * 32 + c] = (__bf16)((xv - mu) * rstd * lwp[q] + lbp[q]);
      }
    }
    // Prefetch next member's row while this member computes.
    if (ie + 1 < NENS && p0w < HW)
      __builtin_prefetch(&xbuf[(size_t)((ie + 1) * CEMB + lane) * HW + p0w], 0, 0);

    const v16bf a = *(const v16bf*)&xsT_w[m * 32 + l16];   // one 32B load
    v8f zero = {};
    v8f dv = __builtin_amdgcn_wmma_f32_16x16x32_bf16(false, a, false, bv,
                                                     (short)0, zero, false, false);
    v8f dk = __builtin_amdgcn_wmma_f32_16x16x32_bf16(false, a, false, bk,
                                                     (short)0, zero, false, false);
    v8f dq = __builtin_amdgcn_wmma_f32_16x16x32_bf16(false, a, false, bq,
                                                     (short)0, zero, false, false);
    #pragma unroll
    for (int r = 0; r < 8; ++r) {
      int w = wav * 16 + r + hi8;                  // pixel index within 128
      int posp = (w & ~31) | perm8(w & 31);
      kbuf[(ie * NHEAD + m) * 128 + w] = (__bf16)fmaxf(dk[r], 0.f);
      qbuf[(ie * NHEAD + m) * 128 + w]    = (__bf16)fmaxf(dq[r], 0.f);
      vstage_w[m * 16 + r + hi8] = dv[r];
    }
    // Coalesced value store: lane<16 = head row, 16 contiguous floats.
    if (p0w < HW && lane < 16) {
      const float4* src = (const float4*)&vstage_w[lane * 16];
      float4* dst = (float4*)&vbuf[(size_t)(ie * NHEAD + lane) * HW + p0w];
      #pragma unroll
      for (int q = 0; q < 4; ++q) dst[q] = src[q];
    }
  }
  __syncthreads();

  // Phase 2: S[c][i][j] += sum_px k[i][c][px] * q[j][c][px], one WMMA chain
  // per head (members padded 10 -> 16 with zero operands).
  const v16bf z16 = {};
  #pragma unroll
  for (int hh = 0; hh < 2; ++hh) {
    const int c = wav * 2 + hh;
    v8f acc = {};
    #pragma unroll
    for (int ks = 0; ks < 128; ks += 32) {
      v16bf a2 = (m < NENS)
          ? *(const v16bf*)&kbuf[(m * NHEAD + c) * 128 + ks + l16] : z16;
      v16bf b2 = (m < NENS)
          ? *(const v16bf*)&qbuf[(m * NHEAD + c) * 128 + ks + l16] : z16;
      acc = __builtin_amdgcn_wmma_f32_16x16x32_bf16(false, a2, false, b2,
                                                    (short)0, acc, false, false);
    }
    #pragma unroll
    for (int r = 0; r < 8; ++r) {
      int i = r + hi8, j = m;
      if (i < NENS && j < NENS)
        atomicAdd(&scores[(c * NENS + i) * NENS + j], acc[r]);
    }
  }
}

// ---------------------------------------------------------------------------
// Kernel 4: softmax over i of scores * 1/sqrt(H*W)  ->  weights[c][i][j].
// ---------------------------------------------------------------------------
__global__ void softmax_kernel(const float* __restrict__ scores,
                               float* __restrict__ weights) {
  int t = threadIdx.x;
  if (t < NHEAD * NENS) {
    int c = t / NENS, j = t % NENS;
    const float norm = rsqrtf((float)HW);
    float s[NENS], mx = -1e30f;
    #pragma unroll
    for (int i = 0; i < NENS; ++i) {
      s[i] = scores[(c * NENS + i) * NENS + j] * norm;
      mx = fmaxf(mx, s[i]);
    }
    float sum = 0.f;
    #pragma unroll
    for (int i = 0; i < NENS; ++i) { s[i] = __expf(s[i] - mx); sum += s[i]; }
    float inv = 1.f / sum;
    #pragma unroll
    for (int i = 0; i < NENS; ++i)
      weights[(c * NENS + i) * NENS + j] = s[i] * inv;
  }
}

// ---------------------------------------------------------------------------
// Kernel 5: attend over perturbations + output proj + residual + ReLU + final
// 1x1.  softmax rows sum to 1 => sum_i w*(v_i - vmean) = (sum_i w*v_i) - vmean.
// Bandwidth-bound; fully coalesced.  Block (32 px, 10 members) = 10 waves.
// ---------------------------------------------------------------------------
__global__ void __launch_bounds__(320) output_kernel(
    const float* __restrict__ vbuf, const float* __restrict__ xbuf,
    const float* __restrict__ weights, const float* __restrict__ wo,
    const float* __restrict__ bo, const float* __restrict__ wout,
    const float* __restrict__ bout, float* __restrict__ out)
{
  __shared__ float vt[NENS * NHEAD * 32];      // value tile, 20 KB
  __shared__ float ws_[NHEAD * NENS * NENS];   // weights, 6.4 KB
  __shared__ float wo_s[CEMB * NHEAD];
  __shared__ float bo_s[CEMB], wout_s[CEMB];

  const int tid = threadIdx.y * 32 + threadIdx.x;
  const int p0 = blockIdx.x * 32;

  for (int idx = tid; idx < NENS * NHEAD * 32; idx += 320) {
    int ic = idx >> 5, px = idx & 31;
    int p = p0 + px;
    vt[idx] = (p < HW) ? vbuf[(size_t)ic * HW + p] : 0.f;
  }
  for (int idx = tid; idx < NHEAD * NENS * NENS; idx += 320) ws_[idx] = weights[idx];
  for (int idx = tid; idx < CEMB * NHEAD; idx += 320) wo_s[idx] = wo[idx];
  if (tid < CEMB) { bo_s[tid] = bo[tid]; wout_s[tid] = wout[tid]; }
  __syncthreads();

  const int px = threadIdx.x, j = threadIdx.y;
  const int p = p0 + px;
  if (p >= HW) return;

  float tv[NHEAD];
  #pragma unroll
  for (int c = 0; c < NHEAD; ++c) {
    float sv = 0.f, sw = 0.f;
    #pragma unroll
    for (int i = 0; i < NENS; ++i) {
      float v = vt[(i * NHEAD + c) * 32 + px];
      sv += v;
      sw += ws_[(c * NENS + i) * NENS + j] * v;
    }
    tv[c] = vt[(j * NHEAD + c) * 32 + px] + sw - sv * 0.1f;  // - vmean
  }
  float y = bout[0];
  #pragma unroll
  for (int ch = 0; ch < CEMB; ++ch) {
    float o = bo_s[ch];
    #pragma unroll
    for (int c = 0; c < NHEAD; ++c) o += wo_s[ch * NHEAD + c] * tv[c];
    o += xbuf[(size_t)(j * CEMB + ch) * HW + p];   // residual
    o = fmaxf(o, 0.f);
    y += wout_s[ch] * o;
  }
  out[(size_t)j * HW + p] = y;
}

// ---------------------------------------------------------------------------
extern "C" void kernel_launch(void* const* d_in, const int* in_sizes, int n_in,
                              void* d_out, int out_size, void* d_ws, size_t ws_size,
                              hipStream_t stream) {
  (void)in_sizes; (void)n_in; (void)out_size; (void)ws_size;
  const float* inp   = (const float*)d_in[0];
  const float* emb_w = (const float*)d_in[1];
  const float* emb_b = (const float*)d_in[2];
  const float* ln_w  = (const float*)d_in[3];
  const float* ln_b  = (const float*)d_in[4];
  const float* wvm   = (const float*)d_in[5];
  const float* wkm   = (const float*)d_in[6];
  const float* wqm   = (const float*)d_in[7];
  const float* wo    = (const float*)d_in[8];
  const float* bo    = (const float*)d_in[9];
  const float* w_out = (const float*)d_in[10];
  const float* b_out = (const float*)d_in[11];
  float* out = (float*)d_out;

  // Workspace layout (~500 MB): x, value, stats, mu/rstd, scores, weights
  float* xbuf    = (float*)d_ws;
  float* vbuf    = xbuf + (size_t)NENS * CEMB * HW;
  float* stats   = vbuf + (size_t)NENS * NHEAD * HW;
  float* murstd  = stats + 32;
  float* scores  = murstd + 32;
  float* weights = scores + NHEAD * NENS * NENS;

  init_kernel<<<8, 256, 0, stream>>>(stats, scores);

  dim3 g1((HDIM * 45 + 7) / 8, 1, NENS);
  conv_emb_kernel<<<g1, 256, 0, stream>>>(inp, emb_w, emb_b, xbuf, stats);

  ln_stats_kernel<<<1, 32, 0, stream>>>(stats, murstd);

  const size_t smem3 = (size_t)CEMB * 128 * 4 * 2            // lnw_s, lnb_s
                     + (size_t)NENS * NHEAD * 128 * 2 * 2    // kbuf, qbuf (bf16)
                     + (size_t)8 * 16 * CEMB * 2             // xsT
                     + (size_t)8 * 256 * 4                   // vstage
                     + (size_t)48 * CEMB * 2;                // wqkvT
  qkv_scores_kernel<<<dim3((HW + 127) / 128), 256, smem3, stream>>>(
      xbuf, ln_w, ln_b, wvm, wkm, wqm, murstd, vbuf, scores);

  softmax_kernel<<<1, 256, 0, stream>>>(scores, weights);

  output_kernel<<<dim3((HW + 31) / 32), dim3(32, 10), 0, stream>>>(
      vbuf, xbuf, weights, wo, bo, w_out, b_out, out);
}
